// MultiScaleGCNDiscriminator_82480551953023
// MI455X (gfx1250) — compile-verified, matrix operand order checked by reading, not run on hardware
//
#include <hip/hip_runtime.h>
#include <hip/hip_bf16.h>

// ---------------- problem constants ----------------
#define NN    50000
#define EEDG  800000
#define E2    (EEDG + NN)   // edges + self loops
#define GGR   64
#define HID   128
#define EPSV  1e-5f

typedef __attribute__((ext_vector_type(16))) _Float16 v16h;
typedef __attribute__((ext_vector_type(8)))  float    v8f;

// ---------------- float <-> monotone uint encoding (for atomic max on float) ---
__device__ __forceinline__ unsigned fenc(float f) {
  unsigned u = __float_as_uint(f);
  return (u & 0x80000000u) ? ~u : (u | 0x80000000u);
}
__device__ __forceinline__ float fdec(unsigned u) {
  return (u & 0x80000000u) ? __uint_as_float(u & 0x7FFFFFFFu) : __uint_as_float(~u);
}

// =====================================================================
// Weight packing: fp32 W[K=128, OUT] -> f16 B-fragment layout.
// B 32x16 f16 fragment per (ktile t, ntile j): lane l holds column
// n = j*16 + (l&15); klo = (l<16)?0:8; half h -> K = t*32 + (h<8 ? klo+h : klo+h+8)
// (mirrors the ISA 16-bit A 16x32 layout with M<->N roles swapped).
// =====================================================================
__global__ void pack_b_kernel(const float* __restrict__ W, _Float16* __restrict__ Bp, int OUT) {
  int total = 128 * OUT;                // 4 ktiles * (OUT/16) ntiles * 32 lanes * 16 halves
  int i = blockIdx.x * blockDim.x + threadIdx.x;
  if (i >= total) return;
  int h    = i & 15;
  int lane = (i >> 4) & 31;
  int tj   = i >> 9;                    // t*ntiles + j
  int ntiles = OUT >> 4;
  int t = tj / ntiles, j = tj - t * ntiles;
  int n   = j * 16 + (lane & 15);
  int klo = (lane < 16) ? 0 : 8;
  int kk  = (h < 8) ? (klo + h) : (klo + h + 8);
  int k   = t * 32 + kk;
  Bp[i] = (_Float16)W[k * OUT + n];
}

// ---- shared helpers for WMMA GEMMs --------------------------------
__device__ __forceinline__ void load_tile16x128(const float* __restrict__ X,
                                                float* tile, int m0, int rows,
                                                int tid, int nthreads) {
  for (int idx = tid; idx < 16 * 32; idx += nthreads) {
    int r = idx >> 5, c4 = idx & 31;
    int row = m0 + r;
    float4 v = make_float4(0.f, 0.f, 0.f, 0.f);
    if (row < rows) v = ((const float4*)X)[row * 32 + c4];
    ((float4*)tile)[idx] = v;
  }
}
__device__ __forceinline__ v16h make_a_frag(const float* tile, int ml, int klo, int t) {
  v16h a;
#pragma unroll
  for (int h = 0; h < 16; ++h) {
    int kk = (h < 8) ? (klo + h) : (klo + h + 8);
    a[h] = (_Float16)tile[ml * 128 + t * 32 + kk];
  }
  return a;
}

// =====================================================================
// WMMA GEMM: Y[rows, OUT] = X[rows,128] @ Wpacked (+bias, optional relu)
// blockDim = 2*OUT (= 32 * OUT/16 lanes), one 16-row tile per block.
// =====================================================================
__global__ void gemm128_wmma(const float* __restrict__ X,
                             const _Float16* __restrict__ Bp,
                             const float* __restrict__ bias,
                             float* __restrict__ Y,
                             int rows, int OUT, int do_relu) {
  __shared__ float tile[16 * 128];
  int m0 = blockIdx.x * 16;
  load_tile16x128(X, tile, m0, rows, threadIdx.x, blockDim.x);
  __syncthreads();

  int wave = threadIdx.x >> 5;     // output ntile
  int lane = threadIdx.x & 31;
  int ml   = lane & 15;
  int klo  = (lane < 16) ? 0 : 8;
  int ntiles = OUT >> 4;

  v8f acc = {0.f, 0.f, 0.f, 0.f, 0.f, 0.f, 0.f, 0.f};
#pragma unroll
  for (int t = 0; t < 4; ++t) {
    v16h a = make_a_frag(tile, ml, klo, t);
    v16h b = *(const v16h*)(Bp + (((t * ntiles + wave) * 32 + lane) << 4));
    acc = __builtin_amdgcn_wmma_f32_16x16x32_f16(false, a, false, b,
                                                 (short)0, acc, false, false);
  }
  int col = wave * 16 + ml;
  float bv = bias ? bias[col] : 0.f;
  int roff = (lane < 16) ? 0 : 8;
#pragma unroll
  for (int r = 0; r < 8; ++r) {
    int row = m0 + r + roff;
    if (row < rows) {
      float v = acc[r] + bv;
      if (do_relu) v = fmaxf(v, 0.f);
      Y[row * OUT + col] = v;
    }
  }
}

// =====================================================================
// Fused dual-output WMMA GEMM (OUT=128 fixed, blockDim=256):
//   Y0 = X @ Bp0 ; Y1 = X @ Bp1
// Builds each A fragment once, issues two WMMAs -> halves X/LDS traffic
// for the per-layer GCN+GAT transform pair (both consume the same x).
// =====================================================================
__global__ void gemm128_wmma_dual(const float* __restrict__ X,
                                  const _Float16* __restrict__ Bp0,
                                  const _Float16* __restrict__ Bp1,
                                  float* __restrict__ Y0,
                                  float* __restrict__ Y1,
                                  int rows) {
  __shared__ float tile[16 * 128];
  int m0 = blockIdx.x * 16;
  load_tile16x128(X, tile, m0, rows, threadIdx.x, blockDim.x);
  __syncthreads();

  int wave = threadIdx.x >> 5;     // output ntile (0..7)
  int lane = threadIdx.x & 31;
  int ml   = lane & 15;
  int klo  = (lane < 16) ? 0 : 8;

  v8f acc0 = {0.f, 0.f, 0.f, 0.f, 0.f, 0.f, 0.f, 0.f};
  v8f acc1 = {0.f, 0.f, 0.f, 0.f, 0.f, 0.f, 0.f, 0.f};
#pragma unroll
  for (int t = 0; t < 4; ++t) {
    v16h a = make_a_frag(tile, ml, klo, t);
    int boff = (((t * 8 + wave) * 32 + lane) << 4);
    v16h b0 = *(const v16h*)(Bp0 + boff);
    v16h b1 = *(const v16h*)(Bp1 + boff);
    acc0 = __builtin_amdgcn_wmma_f32_16x16x32_f16(false, a, false, b0,
                                                  (short)0, acc0, false, false);
    acc1 = __builtin_amdgcn_wmma_f32_16x16x32_f16(false, a, false, b1,
                                                  (short)0, acc1, false, false);
  }
  int col = wave * 16 + ml;
  int roff = (lane < 16) ? 0 : 8;
#pragma unroll
  for (int r = 0; r < 8; ++r) {
    int row = m0 + r + roff;
    if (row < rows) {
      Y0[row * 128 + col] = acc0[r];
      Y1[row * 128 + col] = acc1[r];
    }
  }
}

// =====================================================================
// graph structure kernels
// =====================================================================
__global__ void deg_accum(const int* __restrict__ dst, float* deg, int E) {
  int e = blockIdx.x * blockDim.x + threadIdx.x;
  if (e < E) atomicAdd(&deg[dst[e]], 1.f);
}
__global__ void dis_kernel(const float* __restrict__ deg, float* dis, int n) {
  int i = blockIdx.x * blockDim.x + threadIdx.x;
  if (i < n) dis[i] = rsqrtf(deg[i] + 1.f);   // +1 self loop; always > 0
}
__global__ void cnt_accum(const int* __restrict__ batch, float* cnt, int n) {
  int i = blockIdx.x * blockDim.x + threadIdx.x;
  if (i < n) atomicAdd(&cnt[batch[i]], 1.f);
}

// =====================================================================
// GAT kernels
// =====================================================================
__global__ void gat_scores(const float* __restrict__ hh,
                           const float* __restrict__ asrc,
                           const float* __restrict__ adst,
                           float* __restrict__ a_s, float* __restrict__ a_d, int n) {
  int i = blockIdx.x * blockDim.x + threadIdx.x;
  if (i >= n * 4) return;
  int node = i >> 2, h = i & 3;
  const float* base = hh + node * 128 + h * 32;
  float ss = 0.f, dd = 0.f;
#pragma unroll 8
  for (int c = 0; c < 32; ++c) {
    float v = base[c];
    ss += v * asrc[h * 32 + c];
    dd += v * adst[h * 32 + c];
  }
  a_s[i] = ss; a_d[i] = dd;
}

__global__ void edge_logits(const int* __restrict__ src, const int* __restrict__ dst,
                            const float* __restrict__ a_s, const float* __restrict__ a_d,
                            float* __restrict__ elog, unsigned* __restrict__ emax) {
  int e = blockIdx.x * blockDim.x + threadIdx.x;
  if (e >= E2) return;
  int s = (e < EEDG) ? src[e] : (e - EEDG);
  int d = (e < EEDG) ? dst[e] : (e - EEDG);
#pragma unroll
  for (int h = 0; h < 4; ++h) {
    float v = a_s[s * 4 + h] + a_d[d * 4 + h];
    v = (v > 0.f) ? v : 0.2f * v;               // leaky relu 0.2
    elog[e * 4 + h] = v;
    atomicMax(&emax[d * 4 + h], fenc(v));
  }
}

__global__ void edge_exp(const int* __restrict__ src, const int* __restrict__ dst,
                         const float* __restrict__ elog, const unsigned* __restrict__ emax,
                         float* __restrict__ ee, float* __restrict__ esum) {
  int e = blockIdx.x * blockDim.x + threadIdx.x;
  if (e >= E2) return;
  int d = (e < EEDG) ? dst[e] : (e - EEDG);
#pragma unroll
  for (int h = 0; h < 4; ++h) {
    float v = expf(elog[e * 4 + h] - fdec(emax[d * 4 + h]));
    ee[e * 4 + h] = v;
    atomicAdd(&esum[d * 4 + h], v);
  }
}

__global__ void h_init(const float* __restrict__ gcn_b, const float* __restrict__ gat_b,
                       float* __restrict__ h, int n) {
  int i = blockIdx.x * blockDim.x + threadIdx.x;
  if (i >= n * 128) return;
  int f = i & 127;
  h[i] = gcn_b[f] + gat_b[f];
}

// fused GCN + GAT aggregation, one wave (32 lanes) per edge, 4 feats/lane.
// Random-row gathers dominate -> prefetch the source rows (global_prefetch_b8)
// as soon as s is known, overlapping the fetch with alpha computation.
__global__ void edge_aggr(const int* __restrict__ src, const int* __restrict__ dst,
                          const float* __restrict__ dis,
                          const float* __restrict__ ee, const float* __restrict__ esum,
                          const float* __restrict__ g, const float* __restrict__ hh,
                          float* __restrict__ hout) {
  int gid = blockIdx.x * blockDim.x + threadIdx.x;
  int e = gid >> 5, lane = gid & 31;
  if (e >= E2) return;
  int s = (e < EEDG) ? src[e] : (e - EEDG);
  int d = (e < EEDG) ? dst[e] : (e - EEDG);
  __builtin_prefetch(g + (size_t)s * 128 + lane, 0, 1);
  __builtin_prefetch(hh + (size_t)s * 128 + lane, 0, 1);
  float coef = dis[s] * dis[d];
  float alpha[4];
#pragma unroll
  for (int q = 0; q < 4; ++q) alpha[q] = ee[e * 4 + q] / esum[d * 4 + q];
#pragma unroll
  for (int q = 0; q < 4; ++q) {          // feature f = lane + 32*q, head = q
    int f = lane + (q << 5);
    float v = coef * g[s * 128 + f] + alpha[q] * hh[s * 128 + f];
    atomicAdd(&hout[d * 128 + f], v);
  }
}

// =====================================================================
// GraphNorm kernels
// =====================================================================
__global__ void seg_accum128(const float* __restrict__ h, const int* __restrict__ batch,
                             float* __restrict__ gsum, int n) {
  int i = blockIdx.x * blockDim.x + threadIdx.x;
  if (i >= n * 128) return;
  atomicAdd(&gsum[batch[i >> 7] * 128 + (i & 127)], h[i]);
}
__global__ void seg_div(const float* __restrict__ s, const float* __restrict__ cnt,
                        float* __restrict__ o, int total, int feat) {
  int i = blockIdx.x * blockDim.x + threadIdx.x;
  if (i < total) o[i] = s[i] / fmaxf(cnt[i / feat], 1.f);
}
__global__ void norm_center(const float* __restrict__ h, const int* __restrict__ batch,
                            const float* __restrict__ mean, const float* __restrict__ ms,
                            float* __restrict__ tmp, float* __restrict__ gvar, int n) {
  int i = blockIdx.x * blockDim.x + threadIdx.x;
  if (i >= n * 128) return;
  int f = i & 127;
  float t = h[i] - ms[f] * mean[batch[i >> 7] * 128 + f];
  tmp[i] = t;
  atomicAdd(&gvar[batch[i >> 7] * 128 + f], t * t);
}
__global__ void norm_final(const float* __restrict__ tmp, const int* __restrict__ batch,
                           const float* __restrict__ gvar, const float* __restrict__ w,
                           const float* __restrict__ b, float* __restrict__ x,
                           int n, int first) {
  int i = blockIdx.x * blockDim.x + threadIdx.x;
  if (i >= n * 128) return;
  int f = i & 127;
  float o = tmp[i] * rsqrtf(gvar[batch[i >> 7] * 128 + f] + EPSV);
  float hv = fmaxf(w[f] * o + b[f], 0.f);
  x[i] = first ? hv : (x[i] + hv);
}

// =====================================================================
// pooling / head kernels
// =====================================================================
__global__ void node_deg_accum(const int* __restrict__ src, const int* __restrict__ dst,
                               float* __restrict__ nd, int E) {
  int e = blockIdx.x * blockDim.x + threadIdx.x;
  if (e >= E) return;
  atomicAdd(&nd[src[e]], 1.f);
  atomicAdd(&nd[dst[e]], 1.f);
}
__global__ void seg_max_scalar(const float* __restrict__ v, const int* __restrict__ batch,
                               unsigned* __restrict__ mx, int n) {
  int i = blockIdx.x * blockDim.x + threadIdx.x;
  if (i < n) atomicMax(&mx[batch[i]], fenc(v[i]));
}
__global__ void hub_imp_out(const unsigned* __restrict__ mx, float* __restrict__ out, int g) {
  int i = blockIdx.x * blockDim.x + threadIdx.x;
  if (i < g) out[i] = (fdec(mx[i]) > 0.f) ? 1.f : 0.f;
}
__global__ void dense_small(const float* __restrict__ A, const float* __restrict__ W,
                            const float* __restrict__ b, float* __restrict__ Y,
                            int rows, int K, int OUT, int do_relu) {
  int i = blockIdx.x * blockDim.x + threadIdx.x;
  if (i >= rows * OUT) return;
  int r = i / OUT, o = i - r * OUT;
  float acc = b ? b[o] : 0.f;
  for (int k = 0; k < K; ++k) acc += A[r * K + k] * W[k * OUT + o];
  if (do_relu) acc = fmaxf(acc, 0.f);
  Y[i] = acc;
}
__global__ void seg_accum_feat(const float* __restrict__ v, const int* __restrict__ batch,
                               float* __restrict__ s, int n, int feat) {
  int i = blockIdx.x * blockDim.x + threadIdx.x;
  if (i >= n * feat) return;
  atomicAdd(&s[batch[i / feat] * feat + (i % feat)], v[i]);
}
__global__ void max_scalar_all(const float* __restrict__ v, unsigned* __restrict__ mx, int n) {
  int i = blockIdx.x * blockDim.x + threadIdx.x;
  if (i < n) atomicMax(mx, fenc(v[i]));
}
__global__ void exp_sum(const float* __restrict__ v, const unsigned* __restrict__ mx,
                        float* __restrict__ ew, float* __restrict__ ssum, int n) {
  int i = blockIdx.x * blockDim.x + threadIdx.x;
  if (i >= n) return;
  float e = expf(v[i] - fdec(*mx));
  ew[i] = e;
  atomicAdd(ssum, e);
}
__global__ void weighted_pool(const float* __restrict__ x, const float* __restrict__ ew,
                              const float* __restrict__ ssum, const int* __restrict__ batch,
                              float* __restrict__ gemb, int n) {
  int i = blockIdx.x * blockDim.x + threadIdx.x;
  if (i >= n * 128) return;
  int node = i >> 7;
  atomicAdd(&gemb[batch[node] * 128 + (i & 127)], x[i] * ew[node] / (*ssum));
}
__global__ void pool_mean_max(const float* __restrict__ x, const int* __restrict__ batch,
                              float* __restrict__ msum, unsigned* __restrict__ mmax, int n) {
  int i = blockIdx.x * blockDim.x + threadIdx.x;
  if (i >= n * 128) return;
  int a = batch[i >> 7] * 128 + (i & 127);
  atomicAdd(&msum[a], x[i]);
  atomicMax(&mmax[a], fenc(x[i]));
}
__global__ void build_pe(const float* __restrict__ msum, const unsigned* __restrict__ mmax,
                         const float* __restrict__ cnt, float* __restrict__ pe, int g) {
  int i = blockIdx.x * blockDim.x + threadIdx.x;
  if (i >= g * 256) return;
  int gi = i >> 8, f = i & 255;
  pe[i] = (f < 128) ? (msum[gi * 128 + f] / fmaxf(cnt[gi], 1.f))
                    : fdec(mmax[gi * 128 + (f - 128)]);
}
__global__ void build_comb(const float* __restrict__ gemb, const float* __restrict__ hubp,
                           const float* __restrict__ pat, float* __restrict__ comb, int g) {
  int i = blockIdx.x * blockDim.x + threadIdx.x;
  if (i >= g * 140) return;
  int gi = i / 140, f = i - gi * 140;
  float v;
  if (f < 128)      v = gemb[gi * 128 + f];
  else if (f < 132) v = hubp[gi * 4 + (f - 128)];
  else              v = pat[gi * 8 + (f - 132)];
  comb[i] = v;
}

// =====================================================================
// host launcher
// =====================================================================
static inline int cdiv(long a, long b) { return (int)((a + b - 1) / b); }

extern "C" void kernel_launch(void* const* d_in, const int* in_sizes, int n_in,
                              void* d_out, int out_size, void* d_ws, size_t ws_size,
                              hipStream_t stream) {
  (void)in_sizes; (void)n_in; (void)out_size; (void)ws_size;
  const float* X      = (const float*)d_in[0];
  const int*   eidx   = (const int*)d_in[1];
  const int*   batch  = (const int*)d_in[2];
  const float* projW  = (const float*)d_in[3];
  const float* projB  = (const float*)d_in[4];
  const float* gcnW   = (const float*)d_in[5];
  const float* gcnB   = (const float*)d_in[6];
  const float* gatW   = (const float*)d_in[7];
  const float* gatAs  = (const float*)d_in[8];
  const float* gatAd  = (const float*)d_in[9];
  const float* gatB   = (const float*)d_in[10];
  const float* normW  = (const float*)d_in[11];
  const float* normB  = (const float*)d_in[12];
  const float* normMs = (const float*)d_in[13];
  const float* hubW1  = (const float*)d_in[14];
  const float* hubB1  = (const float*)d_in[15];
  const float* hubW2  = (const float*)d_in[16];
  const float* hubB2  = (const float*)d_in[17];
  const float* attW1  = (const float*)d_in[18];
  const float* attB1  = (const float*)d_in[19];
  const float* attW2  = (const float*)d_in[20];
  const float* attB2  = (const float*)d_in[21];
  const float* patW1  = (const float*)d_in[22];
  const float* patB1  = (const float*)d_in[23];
  const float* patW2  = (const float*)d_in[24];
  const float* patB2  = (const float*)d_in[25];
  const float* patW3  = (const float*)d_in[26];
  const float* patB3  = (const float*)d_in[27];
  const float* clsW1  = (const float*)d_in[28];
  const float* clsB1  = (const float*)d_in[29];
  const float* clsW2  = (const float*)d_in[30];
  const float* clsB2  = (const float*)d_in[31];
  const float* clsW3  = (const float*)d_in[32];
  const float* clsB3  = (const float*)d_in[33];

  const int* src = eidx;
  const int* dst = eidx + EEDG;
  float* out = (float*)d_out;
  // output layout: logits[64*2] | hub_imp[64] | graph_emb[64*128] | hub_pooled[64*4] | pattern[64*8]
  float* out_logits = out;
  float* out_hubimp = out + 128;
  float* out_gemb   = out + 192;
  float* out_hubp   = out + 8384;
  float* out_pat    = out + 8640;

  // ---- workspace carve-up ----
  char* wp = (char*)d_ws;
  auto alloc = [&](size_t bytes) -> void* {
    void* r = (void*)wp;
    wp += (bytes + 255) & ~(size_t)255;
    return r;
  };
  float* xcur  = (float*)alloc((size_t)NN * 128 * 4);
  float* gbuf  = (float*)alloc((size_t)NN * 128 * 4);   // gcn-transformed; reused as norm tmp
  float* hhb   = (float*)alloc((size_t)NN * 128 * 4);   // gat-transformed
  float* hbuf  = (float*)alloc((size_t)NN * 128 * 4);   // aggregated h
  float* elog  = (float*)alloc((size_t)E2 * 4 * 4);     // reused as hub1 [N,64]
  float* eebuf = (float*)alloc((size_t)E2 * 4 * 4);     // reused as att1 [N,64]
  float* deg   = (float*)alloc((size_t)NN * 4);
  float* dis   = (float*)alloc((size_t)NN * 4);
  float* a_s   = (float*)alloc((size_t)NN * 4 * 4);
  float* a_d   = (float*)alloc((size_t)NN * 4 * 4);
  unsigned* emax = (unsigned*)alloc((size_t)NN * 4 * 4);
  float* esum  = (float*)alloc((size_t)NN * 4 * 4);
  float* gsum  = (float*)alloc((size_t)GGR * 128 * 4);
  float* gvar  = (float*)alloc((size_t)GGR * 128 * 4);
  float* gcnt  = (float*)alloc((size_t)GGR * 4);
  float* ndeg  = (float*)alloc((size_t)NN * 4);
  unsigned* maxdeg = (unsigned*)alloc((size_t)GGR * 4);
  unsigned* wmax   = (unsigned*)alloc(4);
  float* wsum  = (float*)alloc(4);
  float* ew    = (float*)alloc((size_t)NN * 4);
  float* hubft = (float*)alloc((size_t)NN * 4 * 4);
  float* wlog  = (float*)alloc((size_t)NN * 4);
  float* hubsum = (float*)alloc((size_t)GGR * 4 * 4);
  float* msum  = (float*)alloc((size_t)GGR * 128 * 4);
  unsigned* mmax = (unsigned*)alloc((size_t)GGR * 128 * 4);
  float* pe    = (float*)alloc((size_t)GGR * 256 * 4);
  float* p1    = (float*)alloc((size_t)GGR * 128 * 4);
  float* p2    = (float*)alloc((size_t)GGR * 64 * 4);
  float* comb  = (float*)alloc((size_t)GGR * 140 * 4);
  float* c1    = (float*)alloc((size_t)GGR * 128 * 4);
  float* c2    = (float*)alloc((size_t)GGR * 64 * 4);
  _Float16* pkProj = (_Float16*)alloc(16384 * 2);
  _Float16* pkGcn0 = (_Float16*)alloc(16384 * 2);
  _Float16* pkGcn1 = (_Float16*)alloc(16384 * 2);
  _Float16* pkGcn2 = (_Float16*)alloc(16384 * 2);
  _Float16* pkGat0 = (_Float16*)alloc(16384 * 2);
  _Float16* pkGat1 = (_Float16*)alloc(16384 * 2);
  _Float16* pkGat2 = (_Float16*)alloc(16384 * 2);
  _Float16* pkHub  = (_Float16*)alloc(8192 * 2);
  _Float16* pkAtt  = (_Float16*)alloc(8192 * 2);
  _Float16* pkGcn[3] = {pkGcn0, pkGcn1, pkGcn2};
  _Float16* pkGat[3] = {pkGat0, pkGat1, pkGat2};
  float* hub1 = elog;    // [N,64] alias (layers done before pooling)
  float* att1 = eebuf;   // [N,64] alias

  const int B = 256;

  // ---- weight packing ----
  pack_b_kernel<<<cdiv(16384, B), B, 0, stream>>>(projW, pkProj, 128);
  for (int i = 0; i < 3; ++i) {
    pack_b_kernel<<<cdiv(16384, B), B, 0, stream>>>(gcnW + i * 16384, pkGcn[i], 128);
    pack_b_kernel<<<cdiv(16384, B), B, 0, stream>>>(gatW + i * 16384, pkGat[i], 128);
  }
  pack_b_kernel<<<cdiv(8192, B), B, 0, stream>>>(hubW1, pkHub, 64);
  pack_b_kernel<<<cdiv(8192, B), B, 0, stream>>>(attW1, pkAtt, 64);

  // ---- degrees / per-graph counts ----
  hipMemsetAsync(deg, 0, (size_t)NN * 4, stream);
  deg_accum<<<cdiv(EEDG, B), B, 0, stream>>>(dst, deg, EEDG);
  dis_kernel<<<cdiv(NN, B), B, 0, stream>>>(deg, dis, NN);
  hipMemsetAsync(gcnt, 0, (size_t)GGR * 4, stream);
  cnt_accum<<<cdiv(NN, B), B, 0, stream>>>(batch, gcnt, NN);

  // ---- input projection ----
  gemm128_wmma<<<cdiv(NN, 16), 256, 0, stream>>>(X, pkProj, projB, xcur, NN, 128, 0);

  // ---- layers ----
  for (int i = 0; i < 3; ++i) {
    // fused GCN+GAT transforms: one X pass, 8 WMMAs per wave
    gemm128_wmma_dual<<<cdiv(NN, 16), 256, 0, stream>>>(xcur, pkGcn[i], pkGat[i],
                                                        gbuf, hhb, NN);
    gat_scores<<<cdiv((long)NN * 4, B), B, 0, stream>>>(hhb, gatAs + i * 128, gatAd + i * 128,
                                                        a_s, a_d, NN);
    hipMemsetAsync(emax, 0, (size_t)NN * 16, stream);
    edge_logits<<<cdiv(E2, B), B, 0, stream>>>(src, dst, a_s, a_d, elog, emax);
    hipMemsetAsync(esum, 0, (size_t)NN * 16, stream);
    edge_exp<<<cdiv(E2, B), B, 0, stream>>>(src, dst, elog, emax, eebuf, esum);
    h_init<<<cdiv((long)NN * 128, B), B, 0, stream>>>(gcnB + i * 128, gatB + i * 128, hbuf, NN);
    edge_aggr<<<cdiv((long)E2 * 32, B), B, 0, stream>>>(src, dst, dis, eebuf, esum, gbuf, hhb, hbuf);
    // GraphNorm
    hipMemsetAsync(gsum, 0, (size_t)GGR * 512, stream);
    seg_accum128<<<cdiv((long)NN * 128, B), B, 0, stream>>>(hbuf, batch, gsum, NN);
    seg_div<<<cdiv(GGR * 128, B), B, 0, stream>>>(gsum, gcnt, gsum, GGR * 128, 128);
    hipMemsetAsync(gvar, 0, (size_t)GGR * 512, stream);
    norm_center<<<cdiv((long)NN * 128, B), B, 0, stream>>>(hbuf, batch, gsum, normMs + i * 128,
                                                           gbuf, gvar, NN);
    seg_div<<<cdiv(GGR * 128, B), B, 0, stream>>>(gvar, gcnt, gvar, GGR * 128, 128);
    norm_final<<<cdiv((long)NN * 128, B), B, 0, stream>>>(gbuf, batch, gvar, normW + i * 128,
                                                          normB + i * 128, xcur, NN, i == 0);
  }

  // ---- hub importance ----
  hipMemsetAsync(ndeg, 0, (size_t)NN * 4, stream);
  node_deg_accum<<<cdiv(EEDG, B), B, 0, stream>>>(src, dst, ndeg, EEDG);
  hipMemsetAsync(maxdeg, 0, (size_t)GGR * 4, stream);
  seg_max_scalar<<<cdiv(NN, B), B, 0, stream>>>(ndeg, batch, maxdeg, NN);
  hub_imp_out<<<1, GGR, 0, stream>>>(maxdeg, out_hubimp, GGR);

  // ---- hub feature pooling ----
  gemm128_wmma<<<cdiv(NN, 16), 128, 0, stream>>>(xcur, pkHub, hubB1, hub1, NN, 64, 1);
  dense_small<<<cdiv((long)NN * 4, B), B, 0, stream>>>(hub1, hubW2, hubB2, hubft, NN, 64, 4, 0);
  hipMemsetAsync(hubsum, 0, (size_t)GGR * 16, stream);
  seg_accum_feat<<<cdiv((long)NN * 4, B), B, 0, stream>>>(hubft, batch, hubsum, NN, 4);
  seg_div<<<cdiv(GGR * 4, B), B, 0, stream>>>(hubsum, gcnt, out_hubp, GGR * 4, 4);

  // ---- global-softmax attention pooling ----
  gemm128_wmma<<<cdiv(NN, 16), 128, 0, stream>>>(xcur, pkAtt, attB1, att1, NN, 64, 1);
  dense_small<<<cdiv(NN, B), B, 0, stream>>>(att1, attW2, attB2, wlog, NN, 64, 1, 0);
  hipMemsetAsync(wmax, 0, 4, stream);
  max_scalar_all<<<cdiv(NN, B), B, 0, stream>>>(wlog, wmax, NN);
  hipMemsetAsync(wsum, 0, 4, stream);
  exp_sum<<<cdiv(NN, B), B, 0, stream>>>(wlog, wmax, ew, wsum, NN);
  hipMemsetAsync(out_gemb, 0, (size_t)GGR * 512, stream);
  weighted_pool<<<cdiv((long)NN * 128, B), B, 0, stream>>>(xcur, ew, wsum, batch, out_gemb, NN);

  // ---- pattern features (mean || max pooling -> MLP) ----
  hipMemsetAsync(msum, 0, (size_t)GGR * 512, stream);
  hipMemsetAsync(mmax, 0, (size_t)GGR * 512, stream);
  pool_mean_max<<<cdiv((long)NN * 128, B), B, 0, stream>>>(xcur, batch, msum, mmax, NN);
  build_pe<<<cdiv(GGR * 256, B), B, 0, stream>>>(msum, mmax, gcnt, pe, GGR);
  dense_small<<<cdiv(GGR * 128, B), B, 0, stream>>>(pe, patW1, patB1, p1, GGR, 256, 128, 1);
  dense_small<<<cdiv(GGR * 64, B), B, 0, stream>>>(p1, patW2, patB2, p2, GGR, 128, 64, 1);
  dense_small<<<cdiv(GGR * 8, B), B, 0, stream>>>(p2, patW3, patB3, out_pat, GGR, 64, 8, 0);

  // ---- classifier ----
  build_comb<<<cdiv(GGR * 140, B), B, 0, stream>>>(out_gemb, out_hubp, out_pat, comb, GGR);
  dense_small<<<cdiv(GGR * 128, B), B, 0, stream>>>(comb, clsW1, clsB1, c1, GGR, 140, 128, 1);
  dense_small<<<cdiv(GGR * 64, B), B, 0, stream>>>(c1, clsW2, clsB2, c2, GGR, 128, 64, 1);
  dense_small<<<cdiv(GGR * 2, B), B, 0, stream>>>(c2, clsW3, clsB3, out_logits, GGR, 64, 2, 0);
}
